// CoordinatedDREAMStack_51608327029265
// MI455X (gfx1250) — matrix operand order repaired and checked by reading, not compiled, not run on hardware
//
#include <hip/hip_runtime.h>
#include <hip/hip_bf16.h>
#include <math.h>

// ---------------- problem constants ----------------
#define BB   32
#define TT   48
#define DIN  128
#define HH   768
#define RR   32
#define LL   3
#define NWG  12          // workgroups in persistent grid (H=768 -> 64 cols each)
#define NTHREADS 256     // 8 wave32 waves per WG

typedef __bf16 bf16_t;
typedef __attribute__((ext_vector_type(16))) __bf16 v16bf;
typedef __attribute__((ext_vector_type(8)))  __bf16 v8bf;
typedef __attribute__((ext_vector_type(8)))  float  v8f;

// ---------------- parameter structs ----------------
struct DreamIn {
  const float *Wp[LL], *bp[LL], *Win[LL], *bi[LL], *We[LL], *be[LL], *V[LL], *eta[LL];
  const float *mh1w[LL], *mh1b[LL], *mh2w[LL], *mh2b[LL];
  const float *Wout, *bout, *x;
};
struct DreamWs {
  unsigned *cnt, *gen;
  bf16_t *Wp_bf[LL], *Win_bf[LL], *We_bf[LL], *V_bf[LL], *mh1_bf[LL], *mh2_bf[LL], *Wout_bf;
  float  *h[LL];  bf16_t *h_bf[LL];
  float  *em[LL], *ev[LL], *smu[LL], *ssig[LL], *avs[LL];
  float  *U[LL], *Ut[LL], *mod[LL];
  float  *error; bf16_t *error_bf;
  float  *kbuf, *xv, *base, *errwe, *fast;
  bf16_t *m_bf, *x0_bf;
  float  *sur, *gain, *dtt, *trig, *coef, *etaeff, *partials;
};

// ---------------- device helpers ----------------
__device__ __forceinline__ void grid_sync(unsigned* cnt, unsigned* gen){
  __threadfence();
  __syncthreads();
  __builtin_amdgcn_s_cluster_barrier();   // CDNA5 cluster barrier (NOP when not clustered)
  if (threadIdx.x == 0){
    unsigned g = __hip_atomic_load(gen, __ATOMIC_RELAXED, __HIP_MEMORY_SCOPE_AGENT);
    unsigned a = __hip_atomic_fetch_add(cnt, 1u, __ATOMIC_ACQ_REL, __HIP_MEMORY_SCOPE_AGENT);
    if (a == (unsigned)(NWG - 1)){
      __hip_atomic_store(cnt, 0u, __ATOMIC_RELAXED, __HIP_MEMORY_SCOPE_AGENT);
      __hip_atomic_store(gen, g + 1u, __ATOMIC_RELEASE, __HIP_MEMORY_SCOPE_AGENT);
    } else {
      while (__hip_atomic_load(gen, __ATOMIC_ACQUIRE, __HIP_MEMORY_SCOPE_AGENT) == g){
        __builtin_amdgcn_s_sleep(2);
      }
    }
  }
  __syncthreads();
  __threadfence();
}

// A fragment: 16x32 bf16 tile of row-major A (lda in elements).
// Per ISA 7.12.2: lane<16 -> row m0+lane, K = {k0..k0+7, k0+16..k0+23};
//                 lane>=16 -> row m0+lane-16, K = {k0+8.., k0+24..}.
__device__ __forceinline__ v16bf loadA(const bf16_t* A, int lda, int m0, int k0, int lane){
  const bf16_t* p = A + (size_t)(m0 + (lane & 15)) * lda + (k0 + ((lane >> 4) << 3));
  v8bf lo = *(const v8bf*)p;          // 16B: K = base..base+7
  v8bf hi = *(const v8bf*)(p + 16);   // 16B: K = base+16..base+23
  return __builtin_shufflevector(lo, hi, 0,1,2,3,4,5,6,7,8,9,10,11,12,13,14,15);
}
// B fragment: 32x16 tile of W stored row-major [N][K] (computing X @ W^T).
// lane<16 -> col n0+lane, K = {k0..k0+7, k0+16..k0+23};
// lane>=16 -> col n0+lane-16, K = {k0+8.., k0+24..}.
__device__ __forceinline__ v16bf loadB(const bf16_t* W, int ldw, int n0, int k0, int lane){
  const bf16_t* p = W + (size_t)(n0 + (lane & 15)) * ldw + (k0 + ((lane >> 4) << 3));
  v8bf lo = *(const v8bf*)p;
  v8bf hi = *(const v8bf*)(p + 16);
  return __builtin_shufflevector(lo, hi, 0,1,2,3,4,5,6,7,8,9,10,11,12,13,14,15);
}

template <typename Epi>
__device__ __forceinline__ void gemm_tile(const bf16_t* A, int lda, const bf16_t* W, int ldw,
                                          int K, int m0, int n0, int lane, Epi epi){
  v8f c = {};
  for (int k0 = 0; k0 < K; k0 += 32){
    v16bf a = loadA(A, lda, m0, k0, lane);
    v16bf b = loadB(W, ldw, n0, k0, lane);
    if (k0 + 64 < K)
      __builtin_prefetch(W + (size_t)(n0 + (lane & 15)) * ldw + k0 + 64, 0, 0);
    c = __builtin_amdgcn_wmma_f32_16x16x32_bf16(false, a, false, b, (short)0, c, false, false);
  }
  // C layout: vgpr j, lane l -> row m0 + ((l>>4)<<3) + j, col n0 + (l&15)
  epi(c, m0 + ((lane >> 4) << 3), n0 + (lane & 15));
}

// tiles strided over all waves in the persistent grid
template <typename Epi>
__device__ __forceinline__ void gemm_all(const bf16_t* A, int lda, const bf16_t* W, int ldw,
                                         int N, int K, Epi epi){
  const int lane = threadIdx.x & 31;
  const int gw   = blockIdx.x * (NTHREADS / 32) + (threadIdx.x >> 5);
  const int nw   = NWG * (NTHREADS / 32);
  const int ntiles = (N >> 4) << 1;                    // 2 M-tiles (M=32)
  for (int t = gw; t < ntiles; t += nw)
    gemm_tile(A, lda, W, ldw, K, (t & 1) << 4, (t >> 1) << 4, lane, epi);
}
// tiles strided over the 8 waves of this WG only
template <typename Epi>
__device__ __forceinline__ void gemm_wg(const bf16_t* A, int lda, const bf16_t* W, int ldw,
                                        int N, int K, Epi epi){
  const int lane = threadIdx.x & 31;
  const int w    = threadIdx.x >> 5;
  const int ntiles = (N >> 4) << 1;
  for (int t = w; t < ntiles; t += NTHREADS / 32)
    gemm_tile(A, lda, W, ldw, K, (t & 1) << 4, (t >> 1) << 4, lane, epi);
}

// ---------------- prologue: weight conversion + state init ----------------
__device__ __forceinline__ void cvt_rng(const float* s, bf16_t* d, size_t n, size_t tid, size_t np){
  for (size_t i = tid; i < n; i += np) d[i] = (bf16_t)s[i];
}
__device__ __forceinline__ void fill_rng(float* d, float v, size_t n, size_t tid, size_t np){
  for (size_t i = tid; i < n; i += np) d[i] = v;
}

__global__ void dream_init(DreamIn in, DreamWs ws){
  size_t tid = (size_t)blockIdx.x * blockDim.x + threadIdx.x;
  size_t np  = (size_t)gridDim.x * blockDim.x;
  if (tid == 0){ *ws.cnt = 0u; *ws.gen = 0u; }
  for (int l = 0; l < LL; ++l){
    int d = l ? HH : DIN;
    cvt_rng(in.Wp[l],  ws.Wp_bf[l],  (size_t)d * HH, tid, np);
    cvt_rng(in.Win[l], ws.Win_bf[l], (size_t)HH * d, tid, np);
    cvt_rng(in.We[l],  ws.We_bf[l],  (size_t)HH * d, tid, np);
    cvt_rng(in.V[l],   ws.V_bf[l],   (size_t)RR * d, tid, np);
    cvt_rng(in.mh1w[l], ws.mh1_bf[l], (size_t)(HH/4) * HH, tid, np);
    cvt_rng(in.mh2w[l], ws.mh2_bf[l], (size_t)HH * (HH/4), tid, np);
    fill_rng(ws.h[l],  0.f, (size_t)BB * HH, tid, np);
    for (size_t i = tid; i < (size_t)BB * HH; i += np) ws.h_bf[l][i] = (bf16_t)0.f;
    fill_rng(ws.em[l],  0.f, (size_t)BB * d, tid, np);
    fill_rng(ws.ev[l],  1.f, (size_t)BB * d, tid, np);
    fill_rng(ws.smu[l], 0.f, BB, tid, np);
    fill_rng(ws.ssig[l],1.f, BB, tid, np);
    fill_rng(ws.avs[l], 0.f, BB, tid, np);
    fill_rng(ws.U[l],  0.f, (size_t)BB * HH * RR, tid, np);
    fill_rng(ws.Ut[l], 0.f, (size_t)BB * HH * RR, tid, np);
    fill_rng(ws.mod[l],0.f, (size_t)BB * HH, tid, np);
  }
  cvt_rng(in.Wout, ws.Wout_bf, (size_t)DIN * HH, tid, np);
}

// ---------------- persistent fused recurrence kernel ----------------
__global__ __launch_bounds__(NTHREADS, 1)
void dream_main(DreamIn in, DreamWs ws, float* out){
  __shared__ float red[NTHREADS];
  __shared__ float red2[NTHREADS];
  __shared__ float bld[BB];
  const int tid  = threadIdx.x;
  const int wg   = blockIdx.x;
  const int gtid = wg * NTHREADS + tid;
  const int gstr = NWG * NTHREADS;

  for (int t = 0; t < TT; ++t){
    // ---- X: stage x_t as bf16 ----
    for (int i = gtid; i < BB * DIN; i += gstr){
      int b = i >> 7, n = i & 127;
      ws.x0_bf[i] = (bf16_t)in.x[((size_t)b * TT + t) * DIN + n];
    }
    grid_sync(ws.cnt, ws.gen);

    for (int l = 0; l < LL; ++l){
      const int d = l ? HH : DIN;
      const float*  xf   = l ? ws.h[l-1]    : (in.x + (size_t)t * DIN);
      const int     xl   = l ? HH           : TT * DIN;
      const bf16_t* xbf  = l ? ws.h_bf[l-1] : ws.x0_bf;
      const int     xbl  = l ? HH           : DIN;

      // ---- A: x_pred = h @ Wp^T + bp ; error = x - x_pred (fused epilogue) ----
      {
        const float* bp = in.bp[l]; const float* xr = xf; const int xld = xl;
        float* err = ws.error; bf16_t* errb = ws.error_bf; const int dd = d;
        gemm_all(ws.h_bf[l], HH, ws.Wp_bf[l], HH, d, HH,
          [=](v8f c, int r0, int col){
            float bpv = bp[col];
            #pragma unroll
            for (int j = 0; j < 8; ++j){
              int r = r0 + j;
              float e = xr[(size_t)r * xld + col] - (c[j] + bpv);
              err[r * dd + col]  = e;
              errb[r * dd + col] = (bf16_t)e;
            }
          });
      }
      grid_sync(ws.cnt, ws.gen);

      // ---- B: surprise stats (wg0) || k = err@V^T (wg1) || xV = x@V^T (wg2) ----
      if (wg == 0){
        const int b = tid >> 3, s = tid & 7;
        const float* em = ws.em[l]; const float* ev = ws.ev[l];
        float acc = 0.f, msum = 0.f;
        for (int j = s; j < d; j += 8){
          float e = ws.error[b * d + j];
          float z = (e - em[b * d + j]) * rsqrtf(ev[b * d + j] + 1e-6f);
          acc += z * z;
        }
        const float* modab = (l < LL - 1) ? ws.mod[l + 1] : nullptr;
        if (modab) for (int j = s; j < HH; j += 8) msum += modab[b * HH + j];
        red[tid] = acc; red2[tid] = msum;
        __syncthreads();
        if (s == 0){
          float sum = 0.f, ms = 0.f;
          for (int q = 0; q < 8; ++q){ sum += red[b * 8 + q]; ms += red2[b * 8 + q]; }
          float enorm = sqrtf(sum / (float)d);
          float smu = 0.99f * ws.smu[l][b] + 0.01f * enorm;
          float ssg = 0.99f * ws.ssig[l][b] + 0.01f * fabsf(enorm - smu);
          ws.smu[l][b] = smu; ws.ssig[l][b] = ssg;
          float sz = (enorm - smu) / (ssg + 1e-6f);
          float s0 = 1.f / (1.f + __expf(-(sz - 0.5f)));
          float mmean = ms / (float)HH;
          float surM = modab ? s0 * (1.f + 0.2f * mmean) : s0;
          ws.gain[b] = 1.f + s0;              // gain uses pre-modulation surprise
          ws.sur[b]  = surM;
          float avs = 0.95f * ws.avs[l][b] + 0.05f * surM;
          ws.avs[l][b] = avs;
          ws.trig[b] = (avs > 0.3f) ? 1.f : 0.f;
          float depth = 1.f + 0.5f * (float)l;
          float tau = fminf(fmaxf(depth / (1.f + surM), 0.01f), 50.f);
          ws.dtt[b] = fminf(fmaxf(0.1f / (tau + 0.1f), 0.01f), 0.5f);
          bld[b] = mmean;
        }
        __syncthreads();
        if (tid == 0){
          float g = 0.f;
          for (int q = 0; q < BB; ++q) g += bld[q];
          g /= (float)BB;
          float eta = *in.eta[l];
          ws.etaeff[0] = (l < LL - 1) ? eta * (1.f + 0.2f * (g - 0.5f)) : eta;
        }
        __syncthreads();
        if (s == 0) ws.coef[b] = 0.1f * ws.etaeff[0] * ws.sur[b];
      } else if (wg == 1){
        float* kb = ws.kbuf;
        gemm_wg(ws.error_bf, d, ws.V_bf[l], d, RR, d,
          [=](v8f c, int r0, int col){
            #pragma unroll
            for (int j = 0; j < 8; ++j) kb[(r0 + j) * RR + col] = c[j];
          });
      } else if (wg == 2){
        float* xv = ws.xv;
        gemm_wg(xbf, xbl, ws.V_bf[l], d, RR, d,
          [=](v8f c, int r0, int col){
            #pragma unroll
            for (int j = 0; j < 8; ++j) xv[(r0 + j) * RR + col] = c[j];
          });
      }
      grid_sync(ws.cnt, ws.gen);

      // ---- C: fast-weight pre-update U = 0.99U + 0.01Ut + coef*tanh(h)*k, partial norms ----
      {
        const int b = tid >> 3, s = tid & 7;
        const int hbase = wg * 64;
        const float cf = ws.coef[b];
        float acc = 0.f;
        float* U = ws.U[l]; const float* Ut = ws.Ut[l]; const float* kb = ws.kbuf;
        for (int hi = 0; hi < 8; ++hi){
          int hh = hbase + s * 8 + hi;
          float th = tanhf(ws.h[l][b * HH + hh]) * cf;
          size_t bse = ((size_t)b * HH + hh) * RR;
          #pragma unroll 4
          for (int r = 0; r < RR; ++r){
            float u = 0.99f * U[bse + r] + 0.01f * Ut[bse + r] + th * kb[b * RR + r];
            U[bse + r] = u;
            acc += u * u;
          }
        }
        red[tid] = acc;
        __syncthreads();
        if (s == 0){
          float sum = 0.f;
          for (int q = 0; q < 8; ++q) sum += red[b * 8 + q];
          ws.partials[wg * BB + b] = sum;
        }
      }
      grid_sync(ws.cnt, ws.gen);

      // ---- D1: normalize U, Ut consolidation, fast = U.xV ; GEMMs base & errWe ----
      {
        const int b = tid >> 3, s = tid & 7;
        float nrm = 0.f;
        for (int q = 0; q < NWG; ++q) nrm += ws.partials[q * BB + b];
        nrm = sqrtf(nrm);
        float sc = fminf(1.f, 1.f / (nrm + 1e-6f));
        float tg = ws.trig[b] * 0.01f;
        const int hbase = wg * 64;
        float* U = ws.U[l]; float* Ut = ws.Ut[l];
        for (int hi = 0; hi < 8; ++hi){
          int hh = hbase + s * 8 + hi;
          size_t bse = ((size_t)b * HH + hh) * RR;
          float facc = 0.f;
          #pragma unroll 4
          for (int r = 0; r < RR; ++r){
            float u = U[bse + r] * sc;
            U[bse + r] = u;
            float ut = Ut[bse + r];
            Ut[bse + r] = ut + tg * (u - ut);
            facc += u * ws.xv[b * RR + r];
          }
          ws.fast[b * HH + hh] = facc;
        }
        { const float* bi = in.bi[l]; float* dst = ws.base;
          gemm_all(xbf, xbl, ws.Win_bf[l], d, HH, d,
            [=](v8f c, int r0, int col){
              float bv = bi[col];
              #pragma unroll
              for (int j = 0; j < 8; ++j) dst[(r0 + j) * HH + col] = c[j] + bv;
            });
        }
        { const float* be = in.be[l]; float* dst = ws.errwe;
          gemm_all(ws.error_bf, d, ws.We_bf[l], d, HH, d,
            [=](v8f c, int r0, int col){
              float bv = be[col];
              #pragma unroll
              for (int j = 0; j < 8; ++j) dst[(r0 + j) * HH + col] = c[j] + bv;
            });
        }
      }
      grid_sync(ws.cnt, ws.gen);

      // ---- D2: LTC blend + error injection + leak; em/ev running stats ----
      {
        float* h = ws.h[l]; bf16_t* hb = ws.h_bf[l];
        for (int i = gtid; i < BB * HH; i += gstr){
          int b = i / HH;
          float hv = h[i];
          float ueff = ws.gain[b] * ws.base[i] + 0.1f * ws.fast[i];
          float dt = ws.dtt[b];
          float hl = (1.f - dt) * hv + dt * tanhf(ueff);
          float hn = hl + ws.errwe[i];
          hn = 0.99f * hn + 0.01f * hv;
          h[i] = hn; hb[i] = (bf16_t)hn;
        }
        float* em = ws.em[l]; float* ev = ws.ev[l];
        for (int i = gtid; i < BB * d; i += gstr){
          float e = ws.error[i];
          float m = 0.9f * em[i] + 0.1f * e;
          em[i] = m;
          float dl = e - m;
          ev[i] = 0.9f * ev[i] + 0.1f * dl * dl;
        }
      }
      grid_sync(ws.cnt, ws.gen);

      // ---- E: modulation head (only layers >= 1 are ever consumed) ----
      if (l > 0){
        { const float* mb = in.mh1b[l]; bf16_t* dst = ws.m_bf;
          gemm_all(ws.h_bf[l], HH, ws.mh1_bf[l], HH, HH/4, HH,
            [=](v8f c, int r0, int col){
              #pragma unroll
              for (int j = 0; j < 8; ++j){
                float v = fmaxf(c[j] + mb[col], 0.f);
                dst[(r0 + j) * (HH/4) + col] = (bf16_t)v;
              }
            });
        }
        grid_sync(ws.cnt, ws.gen);
        { const float* mb = in.mh2b[l]; float* dst = ws.mod[l];
          gemm_all(ws.m_bf, HH/4, ws.mh2_bf[l], HH/4, HH, HH/4,
            [=](v8f c, int r0, int col){
              #pragma unroll
              for (int j = 0; j < 8; ++j){
                float v = c[j] + mb[col];
                dst[(r0 + j) * HH + col] = 1.f / (1.f + __expf(-v));
              }
            });
        }
        grid_sync(ws.cnt, ws.gen);
      }
    } // layers

    // ---- F: out[b,t,:] = h_top @ Wout^T + bout ----
    { const float* bo = in.bout; float* dst = out; const int tt = t;
      gemm_all(ws.h_bf[LL-1], HH, ws.Wout_bf, HH, DIN, HH,
        [=](v8f c, int r0, int col){
          #pragma unroll
          for (int j = 0; j < 8; ++j){
            int b = r0 + j;
            dst[((size_t)b * TT + tt) * DIN + col] = c[j] + bo[col];
          }
        });
    }
  } // timesteps
}

// ---------------- host launcher ----------------
extern "C" void kernel_launch(void* const* d_in, const int* in_sizes, int n_in,
                              void* d_out, int out_size, void* d_ws, size_t ws_size,
                              hipStream_t stream){
  DreamIn in{};
  auto F = [&](int i){ return (const float*)d_in[i]; };
  const bool xfirst = (n_in > 0 && in_sizes[0] == BB * TT * DIN);
  if (xfirst){
    // insertion order: x, [Wp,bp,Win,bi,We,be,V,eta,ph1_w,ph1_b,ln_g,ln_b,ph2_w,ph2_b,
    //                      mh1_w,mh1_b,mh2_w,mh2_b] x3, Wout, bout
    in.x = F(0);
    for (int l = 0; l < LL; ++l){
      int b = 1 + 18 * l;
      in.Wp[l]=F(b+0);  in.bp[l]=F(b+1);  in.Win[l]=F(b+2); in.bi[l]=F(b+3);
      in.We[l]=F(b+4);  in.be[l]=F(b+5);  in.V[l]=F(b+6);   in.eta[l]=F(b+7);
      in.mh1w[l]=F(b+14); in.mh1b[l]=F(b+15); in.mh2w[l]=F(b+16); in.mh2b[l]=F(b+17);
    }
    in.Wout = F(55); in.bout = F(56);
  } else {
    // sorted-key flatten: Wout, bout, [V,We,Win,Wp,be,bi,bp,eta,ln_b,ln_g,
    //                     mh1_b,mh1_w,mh2_b,mh2_w,ph1_b,ph1_w,ph2_b,ph2_w] x3, x
    in.Wout = F(0); in.bout = F(1);
    for (int l = 0; l < LL; ++l){
      int b = 2 + 18 * l;
      in.V[l]=F(b+0);  in.We[l]=F(b+1);  in.Win[l]=F(b+2); in.Wp[l]=F(b+3);
      in.be[l]=F(b+4); in.bi[l]=F(b+5);  in.bp[l]=F(b+6);  in.eta[l]=F(b+7);
      in.mh1b[l]=F(b+10); in.mh1w[l]=F(b+11); in.mh2b[l]=F(b+12); in.mh2w[l]=F(b+13);
    }
    in.x = F(56);
  }

  // bump-allocate workspace (256B aligned chunks)
  char* p = (char*)d_ws;
  auto alloc = [&](size_t bytes)->char*{
    char* r = p; p += (bytes + 255) & ~(size_t)255; return r;
  };
  DreamWs ws{};
  ws.cnt = (unsigned*)alloc(sizeof(unsigned));
  ws.gen = (unsigned*)alloc(sizeof(unsigned));
  for (int l = 0; l < LL; ++l){
    size_t d = l ? HH : DIN;
    ws.Wp_bf[l]  = (bf16_t*)alloc(d * HH * 2);
    ws.Win_bf[l] = (bf16_t*)alloc((size_t)HH * d * 2);
    ws.We_bf[l]  = (bf16_t*)alloc((size_t)HH * d * 2);
    ws.V_bf[l]   = (bf16_t*)alloc((size_t)RR * d * 2);
    ws.mh1_bf[l] = (bf16_t*)alloc((size_t)(HH/4) * HH * 2);
    ws.mh2_bf[l] = (bf16_t*)alloc((size_t)HH * (HH/4) * 2);
    ws.h[l]    = (float*)alloc((size_t)BB * HH * 4);
    ws.h_bf[l] = (bf16_t*)alloc((size_t)BB * HH * 2);
    ws.em[l]   = (float*)alloc((size_t)BB * d * 4);
    ws.ev[l]   = (float*)alloc((size_t)BB * d * 4);
    ws.smu[l]  = (float*)alloc(BB * 4);
    ws.ssig[l] = (float*)alloc(BB * 4);
    ws.avs[l]  = (float*)alloc(BB * 4);
    ws.U[l]    = (float*)alloc((size_t)BB * HH * RR * 4);
    ws.Ut[l]   = (float*)alloc((size_t)BB * HH * RR * 4);
    ws.mod[l]  = (float*)alloc((size_t)BB * HH * 4);
  }
  ws.Wout_bf  = (bf16_t*)alloc((size_t)DIN * HH * 2);
  ws.error    = (float*)alloc((size_t)BB * HH * 4);
  ws.error_bf = (bf16_t*)alloc((size_t)BB * HH * 2);
  ws.kbuf     = (float*)alloc(BB * RR * 4);
  ws.xv       = (float*)alloc(BB * RR * 4);
  ws.base     = (float*)alloc((size_t)BB * HH * 4);
  ws.errwe    = (float*)alloc((size_t)BB * HH * 4);
  ws.fast     = (float*)alloc((size_t)BB * HH * 4);
  ws.m_bf     = (bf16_t*)alloc((size_t)BB * (HH/4) * 2);
  ws.x0_bf    = (bf16_t*)alloc((size_t)BB * DIN * 2);
  ws.sur  = (float*)alloc(BB * 4);  ws.gain = (float*)alloc(BB * 4);
  ws.dtt  = (float*)alloc(BB * 4);  ws.trig = (float*)alloc(BB * 4);
  ws.coef = (float*)alloc(BB * 4);  ws.etaeff = (float*)alloc(4);
  ws.partials = (float*)alloc((size_t)NWG * BB * 4);
  (void)ws_size; (void)out_size;

  dream_init<<<dim3(64), dim3(256), 0, stream>>>(in, ws);
  dream_main<<<dim3(NWG), dim3(NTHREADS), 0, stream>>>(in, ws, (float*)d_out);
}
